// EnCAML_46548855554615
// MI455X (gfx1250) — matrix operand order repaired and checked by reading, not compiled, not run on hardware
//
#include <hip/hip_runtime.h>
#include <hip/hip_bf16.h>

#define BB 4
#define SS 2048
#define EE 100
#define FF 64
#define CC 8922

typedef __attribute__((ext_vector_type(16))) __bf16 v16bf;
typedef __attribute__((ext_vector_type(8)))  float  v8f;
typedef int v4i_vs __attribute__((vector_size(16)));   // matches builtin param pointee

union BF16x16 { v16bf v; uint4 q[2]; };
union BF16x8  { uint4 q; __bf16 e[8]; };

#if defined(__has_builtin)
# if __has_builtin(__builtin_amdgcn_global_load_async_to_lds_b128)
#  define HAVE_ASYNC 1
# endif
#endif
#ifndef HAVE_ASYNC
# define HAVE_ASYNC 0
#endif

static __device__ __forceinline__ v16bf load_v16bf(const __bf16* p0, const __bf16* p1) {
  BF16x16 r;
  r.q[0] = *(const uint4*)p0;
  r.q[1] = *(const uint4*)p1;
  return r.v;
}

static __device__ __forceinline__ v8f wmma_bf16(v16bf a, v16bf b, v8f c) {
  return __builtin_amdgcn_wmma_f32_16x16x32_bf16(false, a, false, b, (short)0, c, false, false);
}

#if HAVE_ASYNC
static __device__ __forceinline__ void async_copy16(const __bf16* g, __bf16* l) {
  __builtin_amdgcn_global_load_async_to_lds_b128(
      (__attribute__((address_space(1))) v4i_vs*)g,
      (__attribute__((address_space(3))) v4i_vs*)l, 0, 0);
}
static __device__ __forceinline__ void async_wait0() {
# if __has_builtin(__builtin_amdgcn_s_wait_asynccnt)
  __builtin_amdgcn_s_wait_asynccnt(0);
# else
  asm volatile("s_wait_asynccnt 0x0" ::: "memory");
# endif
}
#endif

// ---------------------------------------------------------------------------
// Repack conv weights w[F,E,k] (f32) into WMMA B-operand blobs (bf16):
// blob[(tap*4+echunk)*4+ftile][lane][16], lane n holds K = sel*16 + 0..15
// ---------------------------------------------------------------------------
__global__ void wprep_kernel(const float* __restrict__ w3, const float* __restrict__ w5,
                             const float* __restrict__ w7, const float* __restrict__ w9,
                             __bf16* __restrict__ wpack) {
  int br = blockIdx.x;
  int k  = 3 + 2*br;
  const float* w = (br==0)?w3:(br==1)?w5:(br==2)?w7:w9;
  __bf16* dst = wpack + (size_t)8192*br*(br+2);
  int total = k*4*4*32;
  for (int idx = threadIdx.x; idx < total; idx += blockDim.x) {
    int lane = idx & 31, ftile = (idx>>5)&3, ch = (idx>>7)&3, t = idx>>9;
    int sel = lane >> 4, n = lane & 15, f = ftile*16 + n;
    __bf16* o = dst + ((size_t)((t*4+ch)*4+ftile)*32 + lane)*16;
    #pragma unroll
    for (int i = 0; i < 16; ++i) {
      int e = ch*32 + sel*16 + i;
      float v = (e < EE) ? w[((size_t)f*EE + e)*k + t] : 0.0f;
      o[i] = (__bf16)v;
    }
  }
}

// ---------------------------------------------------------------------------
// Embedding gather + conv1d(same) + tanh -> h[branch][b][s][64] bf16.
// Grid (S/32, B, 4); 8 waves = (ss 0..1) x (ftile 0..3); WMMA over taps/E-chunks.
// ---------------------------------------------------------------------------
__global__ void conv_kernel(const int* __restrict__ x, const float* __restrict__ embed,
                            const float* __restrict__ b3, const float* __restrict__ b5,
                            const float* __restrict__ b7, const float* __restrict__ b9,
                            const __bf16* __restrict__ wpack, __bf16* __restrict__ h) {
  __shared__ __bf16 xs[40*128];
  int sblock = blockIdx.x * 32;
  int b = blockIdx.y, br = blockIdx.z;
  int k = 3 + 2*br, hk = k >> 1;
  const float* bias = (br==0)?b3:(br==1)?b5:(br==2)?b7:b9;
  const __bf16* wb = wpack + (size_t)8192*br*(br+2);
  int tid = threadIdx.x;

  for (int i = tid; i < 40*128; i += 256) {
    int r = i >> 7, e = i & 127;
    int s = sblock + r - 4;
    float v = 0.0f;
    if (s >= 0 && s < SS && e < EE) v = embed[(size_t)x[b*SS + s]*EE + e];
    xs[i] = (__bf16)v;
  }
  __syncthreads();

  int wave = tid >> 5, lane = tid & 31;
  int ss = wave & 1, ftile = wave >> 1;
  int sel = lane >> 4, m = lane & 15;

  v8f acc = (v8f)0.0f;
  for (int t = 0; t < k; ++t) {
    int row = ss*16 + m + (t - hk) + 4;         // 0..39
    const __bf16* arow = xs + row*128;
    #pragma unroll
    for (int ch = 0; ch < 4; ++ch) {
      v16bf a = load_v16bf(arow + ch*32 + sel*16, arow + ch*32 + sel*16 + 8);
      v16bf w = ((const BF16x16*)(wb + ((size_t)((t*4+ch)*4+ftile)*32 + lane)*16))->v;
      acc = wmma_bf16(a, w, acc);
    }
  }

  int f = ftile*16 + m;
  float bv = bias[f];
  size_t base = ((size_t)(br*BB + b)*SS)*FF;
  #pragma unroll
  for (int jj = 0; jj < 8; ++jj) {
    int s = sblock + ss*16 + jj + 8*sel;
    float v = acc[jj] + bv;
    v = fminf(fmaxf(v, -15.0f), 15.0f);
    float e2 = __expf(2.0f*v);
    float th = (e2 - 1.0f) / (e2 + 1.0f);       // tanh
    h[base + (size_t)s*FF + f] = (__bf16)th;
  }
}

// ---------------------------------------------------------------------------
// Fused attention, flash-style, double-buffered LDS staging.
// Per 32-s chunk: async copy of raw h chunk to LDS; compute reads A from LDS
// rows, B from thread-transposed hT; one s_barrier per chunk.
// scoresT[s,c]=h·U^T -> exp (lane-local C/D->A repack) -> m[c,f]+=alpha·h.
// ---------------------------------------------------------------------------
__global__ void attn_kernel(const __bf16* __restrict__ h,
                            const float* __restrict__ U3, const float* __restrict__ U5,
                            const float* __restrict__ U7, const float* __restrict__ U9,
                            const float* __restrict__ Wf, const float* __restrict__ bfin,
                            float* __restrict__ yhat) {
  __shared__ __bf16 raw[2][32*FF];              // row-major chunk, 4KB each
  __shared__ __bf16 hT[2][64*40];               // transposed, pitch 40, 5KB each
  int tid = threadIdx.x, b = blockIdx.y;
  int wave = tid >> 5, lane = tid & 31;
  int sel = lane >> 4, m = lane & 15;
  int cbase = blockIdx.x*128 + wave*16;
  int sl = tid >> 3, f8 = (tid & 7)*8;          // staging coords: 16B per thread

  float ya[8] = {0,0,0,0,0,0,0,0};

  for (int br = 0; br < 4; ++br) {
    const float* U = (br==0)?U3:(br==1)?U5:(br==2)?U7:U9;
    int cl = cbase + m; if (cl > CC-1) cl = CC-1;
    BF16x16 Ub0, Ub1;                           // B operands: lane = c col, K = f
    #pragma unroll
    for (int i = 0; i < 16; ++i) {
      Ub0.v[i] = (__bf16)U[(size_t)cl*FF + sel*8 + (i&7) + (i>>3)*16];
      Ub1.v[i] = (__bf16)U[(size_t)cl*FF + 32 + sel*8 + (i&7) + (i>>3)*16];
    }

    const __bf16* hb = h + ((size_t)(br*BB + b)*SS)*FF;
    v8f acc0 = (v8f)0.0f, acc1 = (v8f)0.0f, acc2 = (v8f)0.0f, acc3 = (v8f)0.0f;
    float dsum = 0.0f;

    // ---- prologue: stage chunk 0 into buffer 0
    {
      const __bf16* g = hb + (size_t)sl*FF + f8;
      __bf16* l = &raw[0][sl*FF + f8];
#if HAVE_ASYNC
      async_copy16(g, l);
      async_wait0();
      BF16x8 t8; t8.q = *(const uint4*)l;       // own 16B, just landed
#else
      BF16x8 t8; t8.q = *(const uint4*)g;
      *(uint4*)l = t8.q;
#endif
      #pragma unroll
      for (int j = 0; j < 8; ++j) hT[0][(f8+j)*40 + sl] = t8.e[j];
    }
    __syncthreads();

    for (int sc = 0; sc < SS; sc += 32) {
      int cur = (sc >> 5) & 1, nxt = cur ^ 1;
      bool more = (sc + 32) < SS;

      // issue next chunk's fill (overlaps with this chunk's WMMAs)
#if HAVE_ASYNC
      if (more) async_copy16(hb + (size_t)(sc + 32 + sl)*FF + f8,
                             &raw[nxt][sl*FF + f8]);
#else
      uint4 pay = {0,0,0,0};
      if (more) pay = *(const uint4*)(hb + (size_t)(sc + 32 + sl)*FF + f8);
#endif
      if (sc + 64 < SS) __builtin_prefetch(hb + (size_t)(sc + 64)*FF, 0, 0);

      // ---- score GEMM: A = h rows from LDS, B = U (registers)
      v8f s0 = (v8f)0.0f, s1 = (v8f)0.0f;
      const __bf16* r0 = &raw[cur][(size_t)m*FF];
      const __bf16* r1 = r0 + 16*FF;
      {
        v16bf a;
        a = load_v16bf(r0 + sel*8,      r0 + 16 + sel*8);   s0 = wmma_bf16(a, Ub0.v, s0);
        a = load_v16bf(r0 + 32 + sel*8, r0 + 48 + sel*8);   s0 = wmma_bf16(a, Ub1.v, s0);
        a = load_v16bf(r1 + sel*8,      r1 + 16 + sel*8);   s1 = wmma_bf16(a, Ub0.v, s1);
        a = load_v16bf(r1 + 32 + sel*8, r1 + 48 + sel*8);   s1 = wmma_bf16(a, Ub1.v, s1);
      }

      // exp + denom partials + lane-local C/D->A repack (K = 32 s values)
      BF16x16 alpha;
      #pragma unroll
      for (int jj = 0; jj < 8; ++jj) {
        float p0 = __expf(s0[jj]); dsum += p0; alpha.v[jj]     = (__bf16)p0;
        float p1 = __expf(s1[jj]); dsum += p1; alpha.v[8 + jj] = (__bf16)p1;
      }

      // ---- accumulate m[c,f]; B = h^T rows from LDS (contiguous)
      #pragma unroll
      for (int ft = 0; ft < 4; ++ft) {
        const __bf16* bp = &hT[cur][(ft*16 + m)*40 + sel*8];
        v16bf bv = load_v16bf(bp, bp + 16);
        v8f& ac = (ft==0)?acc0:(ft==1)?acc1:(ft==2)?acc2:acc3;
        ac = wmma_bf16(alpha.v, bv, ac);
      }

      // ---- commit next buffer (own 16B only -> no barrier before transpose)
      if (more) {
#if HAVE_ASYNC
        async_wait0();
        BF16x8 t8; t8.q = *(const uint4*)&raw[nxt][sl*FF + f8];
#else
        BF16x8 t8; t8.q = pay;
        *(uint4*)&raw[nxt][sl*FF + f8] = pay;
#endif
        #pragma unroll
        for (int j = 0; j < 8; ++j) hT[nxt][(f8+j)*40 + sl] = t8.e[j];
        __syncthreads();                        // single barrier per chunk
      }
    }

    // denom per c-col: lanes l and l^16 hold complementary s halves
    dsum += __shfl_xor(dsum, 16, 32);

    #pragma unroll
    for (int jj = 0; jj < 8; ++jj) {
      float dn  = __shfl(dsum, jj + 8*sel, 32);
      float inv = 1.0f / dn;
      int c = cbase + jj + 8*sel;
      float w0 = 0.f, w1 = 0.f, w2 = 0.f, w3v = 0.f;
      if (c < CC) {
        const float* wrow = Wf + (size_t)c*(4*FF) + br*FF + m;
        w0 = wrow[0]; w1 = wrow[16]; w2 = wrow[32]; w3v = wrow[48];
      }
      ya[jj] += inv * (acc0[jj]*w0 + acc1[jj]*w1 + acc2[jj]*w2 + acc3[jj]*w3v);
    }
  }

  #pragma unroll
  for (int jj = 0; jj < 8; ++jj) {
    float v = ya[jj];
    v += __shfl_xor(v, 1, 32);
    v += __shfl_xor(v, 2, 32);
    v += __shfl_xor(v, 4, 32);
    v += __shfl_xor(v, 8, 32);
    int c = cbase + jj + 8*sel;
    if (m == 0 && c < CC) yhat[(size_t)b*CC + c] = v + bfin[c];
  }
}

// ---------------------------------------------------------------------------
// BCE-with-logits mean, single block for determinism. Writes out[B*C].
// ---------------------------------------------------------------------------
__global__ void bce_kernel(const float* __restrict__ yh, const float* __restrict__ y,
                           float* __restrict__ out) {
  __shared__ float red[256];
  int tid = threadIdx.x;
  float s = 0.0f;
  for (int i = tid; i < BB*CC; i += 256) {
    float v = yh[i], t = y[i];
    s += fmaxf(v, 0.0f) - v*t + log1pf(__expf(-fabsf(v)));
  }
  red[tid] = s;
  __syncthreads();
  for (int off = 128; off > 0; off >>= 1) {
    if (tid < off) red[tid] += red[tid + off];
    __syncthreads();
  }
  if (tid == 0) out[BB*CC] = red[0] / (float)(BB*CC);
}

extern "C" void kernel_launch(void* const* d_in, const int* in_sizes, int n_in,
                              void* d_out, int out_size, void* d_ws, size_t ws_size,
                              hipStream_t stream) {
  (void)in_sizes; (void)n_in; (void)out_size; (void)ws_size;
  const int*   x     = (const int*)  d_in[0];
  const float* y     = (const float*)d_in[1];
  const float* embed = (const float*)d_in[2];
  const float* w3 = (const float*)d_in[3];  const float* b3 = (const float*)d_in[4];
  const float* w5 = (const float*)d_in[5];  const float* b5 = (const float*)d_in[6];
  const float* w7 = (const float*)d_in[7];  const float* b7 = (const float*)d_in[8];
  const float* w9 = (const float*)d_in[9];  const float* b9 = (const float*)d_in[10];
  const float* U3 = (const float*)d_in[11]; const float* U5 = (const float*)d_in[12];
  const float* U7 = (const float*)d_in[13]; const float* U9 = (const float*)d_in[14];
  const float* Wf = (const float*)d_in[15]; const float* bf = (const float*)d_in[16];
  float* out = (float*)d_out;

  __bf16* hbuf  = (__bf16*)d_ws;                        // 4*B*S*64 bf16 = 4 MB
  __bf16* wpack = hbuf + (size_t)4*BB*SS*FF;            // 384 KB packed weights

  wprep_kernel<<<dim3(4), 256, 0, stream>>>(w3, w5, w7, w9, wpack);
  conv_kernel<<<dim3(SS/32, BB, 4), 256, 0, stream>>>(x, embed, b3, b5, b7, b9, wpack, hbuf);
  attn_kernel<<<dim3((CC + 127)/128, BB), 256, 0, stream>>>(hbuf, U3, U5, U7, U9, Wf, bf, out);
  bce_kernel<<<1, 256, 0, stream>>>(out, y, out);
}
